// WBP_Decoder_53961969107419
// MI455X (gfx1250) — matrix operand order repaired and checked by reading, not compiled, not run on hardware
//
#include <hip/hip_runtime.h>
#include <hip/hip_bf16.h>
#include <stdint.h>

// WBP LDPC decoder for MI455X (gfx1250, wave32).
// One workgroup (256 thr = 8 waves) per batch row; all edge state in LDS;
// middle MLP layer (32x32) runs on v_wmma_f32_16x16x32_f16.
// A-fragment staging through per-wave LDS with 16B-aligned b128 stores/loads.

typedef __attribute__((ext_vector_type(16))) _Float16 v16h;
typedef __attribute__((ext_vector_type(8)))  float    v8f;

#define EDGES      3072
#define NVARS      1024
#define N_ITER     10
#define CLIPV      (1.0f - 1e-6f)
#define THREADS    256
#define EPW        384     // edges per wave
#define TILES      24      // 16-edge tiles per wave

// LDS layout (floats):
//   8 state arrays of EDGES + W1^T float4 (128) + b1(32) + red(32) + mean(4)
//   + h1 staging (8 waves * 256 dwords)
#define LDS_FLOATS (8*EDGES + 128 + 32 + 32 + 4 + 8*256)
#define LDS_BYTES  (LDS_FLOATS * 4)

__device__ __forceinline__ float elu_f(float x) {
    return x > 0.f ? x : (__expf(x) - 1.f);
}
__device__ __forceinline__ float tanh_half(float x) {   // tanh(0.5*x)
    float xc = fminf(fmaxf(x, -40.f), 40.f);
    float e  = __expf(xc);
    return (e - 1.f) / (e + 1.f);
}
__device__ __forceinline__ float two_atanh(float p) {   // 2*atanh(p)
    return __logf((1.f + p) / (1.f - p));
}

struct alignas(16) uint4v { uint32_t x, y, z, w; };

__global__ __launch_bounds__(THREADS)
void wbp_decoder_kernel(const float* __restrict__ llr,
                        const float* __restrict__ W1, const float* __restrict__ b1,
                        const float* __restrict__ W2, const float* __restrict__ b2,
                        const float* __restrict__ W3, const float* __restrict__ b3,
                        const int* __restrict__ hm_idx,   // (E,2)
                        const int* __restrict__ he_idx,   // (E,5)
                        const int* __restrict__ ro_idx,   // (E,3)
                        const int* __restrict__ vn_edges, // (NVARS)
                        const int* __restrict__ hr_idx,   // (E)
                        float* __restrict__ out) {
    extern __shared__ float smem[];
    float*  s_hr     = smem;
    float*  s_hm     = s_hr     + EDGES;
    float*  s_he     = s_hm     + EDGES;
    float*  s_heres  = s_he     + EDGES;
    float*  s_hmres  = s_heres  + EDGES;
    float*  s_llrres = s_hmres  + EDGES;
    float*  s_llrg   = s_llrres + EDGES;
    float*  s_t      = s_llrg   + EDGES;
    float4* s_W1c4   = (float4*)(s_t + EDGES);  // [32] W1 columns (16B aligned)
    float*  s_b1     = (float*)(s_W1c4 + 32);   // [32]
    float*  s_red    = s_b1 + 32;               // [4][8]
    float*  s_mean   = s_red + 32;              // [4] inverse means
    uint32_t* s_h1   = (uint32_t*)(s_mean + 4); // 8 waves * 256 dwords (16x32 f16 tiles)

    const int tid  = threadIdx.x;
    const int lane = tid & 31;
    const int wv   = tid >> 5;
    const int n0   = lane & 15;           // N column within halftile / M row for A
    const int hi   = lane >> 4;           // lane half-group

    // ---- per-lane constant weight fragments (W2 as two 32x16 B fragments) ----
    v16h bw0, bw1;
#pragma unroll
    for (int idx = 0; idx < 16; ++idx) {
        int K = hi * 16 + idx;            // B layout: lanes 0-15 K=0..15, lanes 16-31 K=16..31
        bw0[idx] = (_Float16)W2[K * 32 + n0];
        bw1[idx] = (_Float16)W2[K * 32 + n0 + 16];
    }
    const float bias2_0 = b2[n0], bias2_1 = b2[n0 + 16];
    const float w3a = W3[n0], w3b = W3[n0 + 16];
    const float bias3 = b3[0];

    if (tid < 32) {
        s_W1c4[tid] = make_float4(W1[0 * 32 + tid], W1[1 * 32 + tid],
                                  W1[2 * 32 + tid], W1[3 * 32 + tid]);
        s_b1[tid] = b1[tid];
    }

    // ---- init per-edge state in LDS ----
    const float* llr_row = llr + (size_t)blockIdx.x * NVARS;
#pragma unroll
    for (int i = 0; i < EDGES / THREADS; ++i) {
        int e = tid + i * THREADS;
        float v = llr_row[hr_idx[e]];
        s_hr[e] = v; s_hm[e] = v; s_llrg[e] = v;
        s_he[e] = 0.f; s_heres[e] = 0.f; s_hmres[e] = 0.f; s_llrres[e] = 0.f;
    }
    __syncthreads();

    uint32_t* h1w = s_h1 + wv * 256;      // this wave's 16x32 f16 staging tile

    for (int it = 0; it < N_ITER; ++it) {
        // ---- phase 1: t = tanh(0.5*h_m) ----
#pragma unroll
        for (int i = 0; i < EDGES / THREADS; ++i) {
            int e = tid + i * THREADS;
            s_t[e] = tanh_half(s_hm[e]);
        }
        __syncthreads();

        // ---- phase 2: check-node update + local sums for the 4 means ----
        float sum0 = 0.f, sum1 = 0.f, sum2 = 0.f, sum3 = 0.f;
#pragma unroll
        for (int i = 0; i < EDGES / THREADS; ++i) {
            int e = tid + i * THREADS;
            const int* g = he_idx + e * 5;
            float p = s_t[g[0]] * s_t[g[1]] * s_t[g[2]] * s_t[g[3]] * s_t[g[4]];
            p = fminf(fmaxf(p, -CLIPV), CLIPV);
            float he_new = two_atanh(p);
            float res = fabsf(he_new - s_he[e]);
            s_he[e] = he_new;
            s_heres[e] = res;
            sum0 += fabsf(he_new);
            sum1 += res;
            sum2 += s_hmres[e];
            sum3 += s_llrres[e];
        }
        // ---- phase 3: block reduction -> inverse means ----
#pragma unroll
        for (int off = 16; off; off >>= 1) {
            sum0 += __shfl_xor(sum0, off);
            sum1 += __shfl_xor(sum1, off);
            sum2 += __shfl_xor(sum2, off);
            sum3 += __shfl_xor(sum3, off);
        }
        if (lane == 0) {
            s_red[wv] = sum0; s_red[8 + wv] = sum1;
            s_red[16 + wv] = sum2; s_red[24 + wv] = sum3;
        }
        __syncthreads();
        if (tid < 4) {
            float s = 0.f;
#pragma unroll
            for (int w = 0; w < 8; ++w) s += s_red[tid * 8 + w];
            float m = s * (1.0f / EDGES);
            s_mean[tid] = (m == 0.f) ? 0.f : 1.f / m;   // where(mean==0, 0, x/mean)
        }
        __syncthreads();
        const float inv0 = s_mean[0], inv1 = s_mean[1], inv2 = s_mean[2], inv3 = s_mean[3];

        // ---- phase 4: per-edge MLP; middle layer on WMMA ----
        for (int tile = 0; tile < TILES; ++tile) {
            const int eb = wv * EPW + tile * 16;
            const int e  = eb + n0;
            float f0 = fabsf(s_he[e]) * inv0;
            float f1 = s_heres[e]  * inv1;
            float f2 = s_hmres[e]  * inv2;
            float f3 = s_llrres[e] * inv3;
            // layer 1: lane computes k = hi*16 .. hi*16+15 for edge row n0,
            // packed as 8 f16-pair dwords -> two b128 stores into the staging tile.
            uint32_t pk[8];
#pragma unroll
            for (int kk = 0; kk < 16; kk += 2) {
                int k = hi * 16 + kk;
                float4 wa = s_W1c4[k];
                float4 wb = s_W1c4[k + 1];
                float a0 = s_b1[k]     + f0 * wa.x + f1 * wa.y + f2 * wa.z + f3 * wa.w;
                float a1 = s_b1[k + 1] + f0 * wb.x + f1 * wb.y + f2 * wb.z + f3 * wb.w;
                union { _Float16 h[2]; uint32_t u; } upk;
                upk.h[0] = (_Float16)elu_f(a0);
                upk.h[1] = (_Float16)elu_f(a1);
                pk[kk >> 1] = upk.u;
            }
            {
                uint4v q0 = { pk[0], pk[1], pk[2], pk[3] };
                uint4v q1 = { pk[4], pk[5], pk[6], pk[7] };
                *(uint4v*)(h1w + n0 * 16 + hi * 8)     = q0;
                *(uint4v*)(h1w + n0 * 16 + hi * 8 + 4) = q1;
            }
            // A fragment (16-bit 16x32 layout, M = lane&15): two b128 loads
            union { v16h v; uint4v q[2]; } af;
            af.q[0] = *(const uint4v*)(h1w + n0 * 16 + hi * 8);
            af.q[1] = *(const uint4v*)(h1w + n0 * 16 + hi * 8 + 4);

            v8f c0, c1;
#pragma unroll
            for (int r = 0; r < 8; ++r) { c0[r] = bias2_0; c1[r] = bias2_1; }
            v8f d0 = __builtin_amdgcn_wmma_f32_16x16x32_f16(false, af.v, false, bw0,
                                                            (short)0, c0, false, false);
            v8f d1 = __builtin_amdgcn_wmma_f32_16x16x32_f16(false, af.v, false, bw1,
                                                            (short)0, c1, false, false);
            // layer 3: elu + dot with W3, reduce over N across the 16-lane group
            float wrow[8];
#pragma unroll
            for (int r = 0; r < 8; ++r) {
                float p = elu_f(d0[r]) * w3a + elu_f(d1[r]) * w3b;
                p += __shfl_xor(p, 1);
                p += __shfl_xor(p, 2);
                p += __shfl_xor(p, 4);
                p += __shfl_xor(p, 8);
                wrow[r] = elu_f(p + bias3);
            }
            // lanes 0-15 hold rows 0..7, lanes 16-31 rows 8..15; one lane per group scales h_e
            if (n0 == 0) {
                int base = eb + hi * 8;
#pragma unroll
                for (int r = 0; r < 8; ++r) s_he[base + r] *= wrow[r];
            }
        }
        __syncthreads();

        // ---- phase 5: variable-node update + residuals ----
#pragma unroll
        for (int i = 0; i < EDGES / THREADS; ++i) {
            int e = tid + i * THREADS;
            const int* g2 = hm_idx + e * 2;
            const int* g3 = ro_idx + e * 3;
            float hr = s_hr[e];
            float hmn = s_he[g2[0]] + s_he[g2[1]] + hr;
            s_hmres[e] = fabsf(hmn - s_hm[e]);
            s_hm[e] = hmn;
            float ln = s_he[g3[0]] + s_he[g3[1]] + s_he[g3[2]] + hr;
            s_llrres[e] = fabsf(ln - s_llrg[e]);
            s_llrg[e] = ln;
        }
        __syncthreads();
    }

    // ---- output: llr_g gathered at vn_edges ----
    float* out_row = out + (size_t)blockIdx.x * NVARS;
#pragma unroll
    for (int i = 0; i < NVARS / THREADS; ++i) {
        int v = tid + i * THREADS;
        out_row[v] = s_llrg[vn_edges[v]];
    }
}

extern "C" void kernel_launch(void* const* d_in, const int* in_sizes, int n_in,
                              void* d_out, int out_size, void* d_ws, size_t ws_size,
                              hipStream_t stream) {
    const float* llr = (const float*)d_in[0];
    const float* W1  = (const float*)d_in[1];
    const float* b1  = (const float*)d_in[2];
    const float* W2  = (const float*)d_in[3];
    const float* b2  = (const float*)d_in[4];
    const float* W3  = (const float*)d_in[5];
    const float* b3  = (const float*)d_in[6];
    const int* hm_idx   = (const int*)d_in[7];
    const int* he_idx   = (const int*)d_in[8];
    const int* ro_idx   = (const int*)d_in[9];
    const int* vn_edges = (const int*)d_in[10];
    const int* hr_idx   = (const int*)d_in[11];
    float* out = (float*)d_out;

    const int B = in_sizes[0] / NVARS;  // 2048
    hipLaunchKernelGGL(wbp_decoder_kernel, dim3(B), dim3(THREADS), LDS_BYTES, stream,
                       llr, W1, b1, W2, b2, W3, b3,
                       hm_idx, he_idx, ro_idx, vn_edges, hr_idx, out);
}
